// MultiHeadAttentionWithRope_41583873360268
// MI455X (gfx1250) — compile-verified, matrix operand order checked by reading, not run on hardware
//
#include <hip/hip_runtime.h>

// ---------------------------------------------------------------------------
// MHA + RoPE for MI455X (gfx1250, wave32).
// All matmuls on v_wmma_f32_16x16x32_bf16 (f32 accumulate); flash attention
// with online softmax in f32; P-tile relayout via LDS.
// Workspace layout (bytes):
//   [ 0M..8M)   xb      : x in bf16                 [4096][1024]
//   [ 8M..10M)  Wqb     : Wq bf16 [e][d]
//   [10M..12M)  Wkb
//   [12M..14M)  Wvb
//   [14M..16M)  Wob
//   [16M..24M)  Qb      : Q bf16 [s][1024] (RoPE'd in place)
//   [24M..32M)  Kb      : K bf16 [s][1024] (RoPE'd in place)
//   [32M..40M)  Vb      : V bf16 [s][1024]
//   [40M..48M)  Vt      : V transposed bf16 [h][64][4096]
//   [48M..56M)  attnb   : attention output bf16 [s][1024]
// ---------------------------------------------------------------------------

#define S_LEN 4096
#define D_MODEL 1024
#define NHEAD 16
#define DKH 64

typedef __attribute__((ext_vector_type(16))) __bf16 v16bf;
typedef __attribute__((ext_vector_type(8)))  __bf16 v8bf;
typedef __attribute__((ext_vector_type(8)))  float  v8f;

// ---------------------------------------------------------------------------
// Fragment loader for 16-bit WMMA operands (A: row=M, B: row=N), source
// row-major [row][k] with contiguous K.
// ISA layout (16-bit 16x32 operand): lanes 0-15 = rows, lanes 16-31 = K+8:
//   per lane: elements base[g*8 .. g*8+7] and base[16+g*8 .. 16+g*8+7].
// Requires (row0*ld + k0) to be a multiple of 8 elements (16B alignment).
// ---------------------------------------------------------------------------
__device__ inline v16bf load_frag(const __bf16* __restrict__ base,
                                  int row0, int k0, int ld) {
  const int lane = threadIdx.x & 31;
  const int r = row0 + (lane & 15);
  const int g = lane >> 4;
  const __bf16* p = base + (size_t)r * ld + k0 + g * 8;
  v8bf lo = *(const v8bf*)(p);
  v8bf hi = *(const v8bf*)(p + 16);
  return __builtin_shufflevector(lo, hi,
      0, 1, 2, 3, 4, 5, 6, 7, 8, 9, 10, 11, 12, 13, 14, 15);
}

__device__ inline v8f wmma_bf16(v16bf a, v16bf b, v8f c) {
  return __builtin_amdgcn_wmma_f32_16x16x32_bf16(
      /*neg_a=*/false, a, /*neg_b=*/false, b,
      /*c_mod=*/(short)0, c, /*reuse_a=*/false, /*reuse_b=*/false);
}

// 16-lane (half-wave) reductions: xor offsets 1,2,4,8 never cross the
// 16-lane boundary, matching the C/D layout groups.
__device__ inline float rmax16(float x) {
  x = fmaxf(x, __shfl_xor(x, 1));
  x = fmaxf(x, __shfl_xor(x, 2));
  x = fmaxf(x, __shfl_xor(x, 4));
  x = fmaxf(x, __shfl_xor(x, 8));
  return x;
}
__device__ inline float rsum16(float x) {
  x += __shfl_xor(x, 1);
  x += __shfl_xor(x, 2);
  x += __shfl_xor(x, 4);
  x += __shfl_xor(x, 8);
  return x;
}

// ---------------------------------------------------------------------------
// fp32 -> bf16 convert
// ---------------------------------------------------------------------------
__global__ void cvt_kernel(const float* __restrict__ in,
                           __bf16* __restrict__ out, int n) {
  int id = blockIdx.x * blockDim.x + threadIdx.x;
  if (id < n) out[id] = (__bf16)in[id];
}

// ---------------------------------------------------------------------------
// GEMM: C[M,N] = A[M,K] * B^T where Bm is stored [N][K] row-major.
// Block = 256 threads = 8 waves arranged 2(M) x 4(N); block tile 128 x 128.
// Each wave computes a 64x32 tile (4x2 WMMA tiles): 8 wmma per 6 fragment
// loads per K-step -> matrix-issue bound, not load bound.
// ---------------------------------------------------------------------------
template <bool OUT_BF16>
__global__ __launch_bounds__(256)
void gemm_kernel(const __bf16* __restrict__ A, const __bf16* __restrict__ Bm,
                 float* __restrict__ Cf, __bf16* __restrict__ Cb,
                 int M, int N, int K) {
  const int w = threadIdx.x >> 5;
  const int lane = threadIdx.x & 31;
  const int m0 = blockIdx.x * 128 + (w >> 2) * 64;
  const int n0 = blockIdx.y * 128 + (w & 3) * 32;

  v8f acc[4][2] = {};
  for (int k0 = 0; k0 < K; k0 += 32) {
    // speculative prefetch of next K tile (global_prefetch_b8)
    if (k0 + 32 < K) {
      __builtin_prefetch(A + (size_t)(m0 + (lane & 15)) * K + k0 + 32, 0, 0);
      __builtin_prefetch(Bm + (size_t)(n0 + (lane & 15)) * K + k0 + 32, 0, 0);
    }
    v16bf b0 = load_frag(Bm, n0, k0, K);
    v16bf b1 = load_frag(Bm, n0 + 16, k0, K);
#pragma unroll
    for (int mi = 0; mi < 4; ++mi) {
      v16bf a = load_frag(A, m0 + mi * 16, k0, K);
      acc[mi][0] = wmma_bf16(a, b0, acc[mi][0]);
      acc[mi][1] = wmma_bf16(a, b1, acc[mi][1]);
    }
  }

  const int col = lane & 15;
  const int g = lane >> 4;
#pragma unroll
  for (int mi = 0; mi < 4; ++mi)
#pragma unroll
    for (int ni = 0; ni < 2; ++ni)
#pragma unroll
      for (int v = 0; v < 8; ++v) {
        int m = m0 + mi * 16 + v + 8 * g;
        int n = n0 + ni * 16 + col;
        if constexpr (OUT_BF16)
          Cb[(size_t)m * N + n] = (__bf16)acc[mi][ni][v];
        else
          Cf[(size_t)m * N + n] = acc[mi][ni][v];
      }
}

// ---------------------------------------------------------------------------
// RoPE in place on bf16 [S][1024]; one thread per (even,odd) pair.
// ---------------------------------------------------------------------------
__global__ void rope_kernel(__bf16* __restrict__ T,
                            const int* __restrict__ pos, int npairs) {
  int id = blockIdx.x * blockDim.x + threadIdx.x;
  if (id >= npairs) return;
  int s = id >> 9;                 // 512 pairs per token
  int rem = id & 511;
  int i = rem & 31;                // pair index within head (0..31)
  int col = ((rem >> 5) << 6) + (i << 1);  // h*64 + 2i
  float p = (float)pos[s];
  float freq = __powf(10000.0f, -((float)(2 * i)) / 64.0f);
  float ang = p * freq;
  float sn, cs;
  __sincosf(ang, &sn, &cs);
  __bf16* ptr = T + (size_t)s * D_MODEL + col;
  float t1 = (float)ptr[0];
  float t2 = (float)ptr[1];
  ptr[0] = (__bf16)(t1 * cs - t2 * sn);
  ptr[1] = (__bf16)(t1 * sn + t2 * cs);
}

// ---------------------------------------------------------------------------
// V transpose: Vt[h][d][s] = Vb[s][h*64+d]   (i.e. Vt[c][s] = Vb[s][c])
// ---------------------------------------------------------------------------
__global__ void vtrans_kernel(const __bf16* __restrict__ Vb,
                              __bf16* __restrict__ Vt) {
  int id = blockIdx.x * blockDim.x + threadIdx.x;
  if (id >= S_LEN * D_MODEL) return;
  int s = id >> 10;
  int c = id & 1023;
  Vt[(size_t)c * S_LEN + s] = Vb[id];
}

// ---------------------------------------------------------------------------
// Flash attention: one wave per (head, 16-query block), key blocks of 32.
// ---------------------------------------------------------------------------
__global__ __launch_bounds__(256)
void attn_kernel(const __bf16* __restrict__ Qb, const __bf16* __restrict__ Kb,
                 const __bf16* __restrict__ Vt, __bf16* __restrict__ attnb) {
  __shared__ __bf16 ldsP[8][16 * 40];  // per-wave 16x32 P tile, stride 40 (pad)

  const int w = threadIdx.x >> 5;
  const int lane = threadIdx.x & 31;
  const int col = lane & 15;
  const int g = lane >> 4;

  const int wg = blockIdx.x * 8 + w;     // 0..4095
  const int h = wg >> 8;                 // head
  const int q0 = (wg & 255) << 4;        // query block start

  // Q fragments (16 x 64), loaded once.
  v16bf qa0 = load_frag(Qb, q0, h * DKH, D_MODEL);
  v16bf qa1 = load_frag(Qb, q0, h * DKH + 32, D_MODEL);
  const __bf16* Vh = Vt + (size_t)h * DKH * S_LEN;

  v8f o[4] = {};
  float m_run[8], l_run[8];
#pragma unroll
  for (int v = 0; v < 8; ++v) { m_run[v] = -1e30f; l_run[v] = 0.0f; }

  for (int j0 = 0; j0 <= q0 + 15; j0 += 32) {
    // ---- S = Q K^T, two 16x16 tiles ----
    v8f sacc[2] = {};
#pragma unroll
    for (int t = 0; t < 2; ++t) {
      v16bf kb0 = load_frag(Kb, j0 + t * 16, h * DKH, D_MODEL);
      v16bf kb1 = load_frag(Kb, j0 + t * 16, h * DKH + 32, D_MODEL);
      sacc[t] = wmma_bf16(qa0, kb0, sacc[t]);
      sacc[t] = wmma_bf16(qa1, kb1, sacc[t]);
    }

    // ---- scale + causal mask ----
    float pv[2][8];
#pragma unroll
    for (int t = 0; t < 2; ++t)
#pragma unroll
      for (int v = 0; v < 8; ++v) {
        int key = j0 + t * 16 + col;
        int qrow = q0 + v + 8 * g;
        float val = sacc[t][v] * 0.125f;  // 1/sqrt(64)
        pv[t][v] = (key <= qrow) ? val : -1e30f;
      }

    // ---- online softmax ----
    float alpha[8];
#pragma unroll
    for (int v = 0; v < 8; ++v) {
      float rm = rmax16(fmaxf(pv[0][v], pv[1][v]));
      float mn = fmaxf(m_run[v], rm);
      float al = __expf(m_run[v] - mn);
      m_run[v] = mn;
      float p0 = __expf(pv[0][v] - mn);
      float p1 = __expf(pv[1][v] - mn);
      pv[0][v] = p0;
      pv[1][v] = p1;
      float rs = rsum16(p0 + p1);
      l_run[v] = l_run[v] * al + rs;
      alpha[v] = al;
    }
#pragma unroll
    for (int t2 = 0; t2 < 4; ++t2)
#pragma unroll
      for (int v = 0; v < 8; ++v) o[t2][v] *= alpha[v];

    // ---- stage P (16x32) to LDS to re-layout as A fragment ----
    __bf16* myP = &ldsP[w][0];
#pragma unroll
    for (int t = 0; t < 2; ++t)
#pragma unroll
      for (int v = 0; v < 8; ++v)
        myP[(v + 8 * g) * 40 + t * 16 + col] = (__bf16)pv[t][v];
    asm volatile("s_wait_dscnt 0" ::: "memory");

    v16bf pa = load_frag(myP, 0, 0, 40);

    // ---- O += P @ V  (B fragments from Vt: row = d, k = key) ----
#pragma unroll
    for (int t2 = 0; t2 < 4; ++t2) {
      v16bf vb = load_frag(Vh, t2 * 16, j0, S_LEN);
      o[t2] = wmma_bf16(pa, vb, o[t2]);
    }
  }

  // ---- normalize and store bf16 ----
#pragma unroll
  for (int t2 = 0; t2 < 4; ++t2)
#pragma unroll
    for (int v = 0; v < 8; ++v) {
      int m = v + 8 * g;
      int d = t2 * 16 + col;
      attnb[(size_t)(q0 + m) * D_MODEL + h * DKH + d] =
          (__bf16)(o[t2][v] / l_run[v]);
    }
}

// ---------------------------------------------------------------------------
// Host launcher
// ---------------------------------------------------------------------------
extern "C" void kernel_launch(void* const* d_in, const int* in_sizes, int n_in,
                              void* d_out, int out_size, void* d_ws,
                              size_t ws_size, hipStream_t stream) {
  const float* x  = (const float*)d_in[0];
  const int*   tp = (const int*)d_in[1];
  const float* Wq = (const float*)d_in[2];
  const float* Wk = (const float*)d_in[3];
  const float* Wv = (const float*)d_in[4];
  const float* Wo = (const float*)d_in[5];
  float* out = (float*)d_out;

  char* ws = (char*)d_ws;
  const size_t MB = 1u << 20;
  __bf16* xb    = (__bf16*)(ws + 0 * MB);
  __bf16* Wqb   = (__bf16*)(ws + 8 * MB);
  __bf16* Wkb   = (__bf16*)(ws + 10 * MB);
  __bf16* Wvb   = (__bf16*)(ws + 12 * MB);
  __bf16* Wob   = (__bf16*)(ws + 14 * MB);
  __bf16* Qb    = (__bf16*)(ws + 16 * MB);
  __bf16* Kb    = (__bf16*)(ws + 24 * MB);
  __bf16* Vb    = (__bf16*)(ws + 32 * MB);
  __bf16* Vt    = (__bf16*)(ws + 40 * MB);
  __bf16* attnb = (__bf16*)(ws + 48 * MB);

  const int nX = S_LEN * D_MODEL;      // 4,194,304
  const int nW = D_MODEL * D_MODEL;    // 1,048,576

  cvt_kernel<<<(nX + 255) / 256, 256, 0, stream>>>(x, xb, nX);
  cvt_kernel<<<(nW + 255) / 256, 256, 0, stream>>>(Wq, Wqb, nW);
  cvt_kernel<<<(nW + 255) / 256, 256, 0, stream>>>(Wk, Wkb, nW);
  cvt_kernel<<<(nW + 255) / 256, 256, 0, stream>>>(Wv, Wvb, nW);
  cvt_kernel<<<(nW + 255) / 256, 256, 0, stream>>>(Wo, Wob, nW);

  dim3 ggrid(S_LEN / 128, D_MODEL / 128);  // (32, 8)
  gemm_kernel<true><<<ggrid, 256, 0, stream>>>(xb, Wqb, nullptr, Qb,
                                               S_LEN, D_MODEL, D_MODEL);
  gemm_kernel<true><<<ggrid, 256, 0, stream>>>(xb, Wkb, nullptr, Kb,
                                               S_LEN, D_MODEL, D_MODEL);
  gemm_kernel<true><<<ggrid, 256, 0, stream>>>(xb, Wvb, nullptr, Vb,
                                               S_LEN, D_MODEL, D_MODEL);

  const int npairs = S_LEN * D_MODEL / 2;
  rope_kernel<<<(npairs + 255) / 256, 256, 0, stream>>>(Qb, tp, npairs);
  rope_kernel<<<(npairs + 255) / 256, 256, 0, stream>>>(Kb, tp, npairs);

  vtrans_kernel<<<(nX + 255) / 256, 256, 0, stream>>>(Vb, Vt);

  attn_kernel<<<512, 256, 0, stream>>>(Qb, Kb, Vt, attnb);

  gemm_kernel<false><<<ggrid, 256, 0, stream>>>(attnb, Wob, out, nullptr,
                                                S_LEN, D_MODEL, D_MODEL);
}
